// GRU_Decoder_19971597927080
// MI455X (gfx1250) — matrix operand order repaired and checked by reading, not compiled, not run on hardware
//
#include <hip/hip_runtime.h>
#include <math.h>

typedef __bf16 bf16;
typedef __attribute__((ext_vector_type(16))) __bf16 v16bf;
typedef __attribute__((ext_vector_type(8)))  __bf16 v8bf;
typedef __attribute__((ext_vector_type(8)))  float  v8f;

#define HID   1024
#define BATCH 128
#define SEQ   512
#define NBLK  64
#define TPB   256   // 8 wave32s -> 64*8 = 512 waves = one 16x16 h-tile per wave
#define HELEM ((size_t)BATCH * HID)

// ---------------------------------------------------------------------------
// Branch-free activations on the recurrence critical path.
// CDNA5 has native V_TANH_F32; sigmoid(x) = 0.5*tanh(x/2) + 0.5.
// ---------------------------------------------------------------------------
__device__ __forceinline__ float fast_tanh(float x) {
#if __has_builtin(__builtin_amdgcn_tanhf)
  return __builtin_amdgcn_tanhf(x);
#elif __has_builtin(__builtin_amdgcn_tanh_f32)
  return __builtin_amdgcn_tanh_f32(x);
#else
  // branch-free fallback: clamp so exp stays finite, then (t-1)/(t+1) via rcp
  float xc = fminf(fmaxf(x, -15.0f), 15.0f);
  float t  = __expf(2.0f * xc);
  return (t - 1.0f) * __builtin_amdgcn_rcpf(t + 1.0f);
#endif
}
__device__ __forceinline__ float fast_sigmoid(float x) {
  return 0.5f * fast_tanh(0.5f * x) + 0.5f;
}

// ---------------------------------------------------------------------------
// fp32 -> bf16 conversion, 8 elements/thread (n must be a multiple of 8)
// ---------------------------------------------------------------------------
__global__ void cvt_f32_to_bf16_v8(const float* __restrict__ src,
                                   bf16* __restrict__ dst, int n8) {
  int i = blockIdx.x * blockDim.x + threadIdx.x;
  if (i < n8) {
    const float4* s = (const float4*)src + 2 * (size_t)i;
    float4 a = s[0], b = s[1];
    v8bf o;
    o[0] = (bf16)a.x; o[1] = (bf16)a.y; o[2] = (bf16)a.z; o[3] = (bf16)a.w;
    o[4] = (bf16)b.x; o[5] = (bf16)b.y; o[6] = (bf16)b.z; o[7] = (bf16)b.w;
    ((v8bf*)dst)[i] = o;
  }
}

// ---------------------------------------------------------------------------
// Device-wide barrier (counter + generation). All NBLK blocks are resident.
// ---------------------------------------------------------------------------
__device__ __forceinline__ void grid_barrier(unsigned* bar) {
  __syncthreads();
  if (threadIdx.x == 0) {
    __threadfence();  // release h-state writes to device scope
    unsigned gen  = __atomic_load_n(bar + 1, __ATOMIC_RELAXED);
    unsigned prev = __atomic_fetch_add(bar, 1u, __ATOMIC_ACQ_REL);
    if (prev == gridDim.x - 1) {
      __atomic_store_n(bar, 0u, __ATOMIC_RELAXED);
      __atomic_store_n(bar + 1, gen + 1u, __ATOMIC_RELEASE);
    } else {
      while (__atomic_load_n(bar + 1, __ATOMIC_ACQUIRE) == gen) {
        __builtin_amdgcn_s_sleep(1);
      }
    }
  }
  __syncthreads();
  __threadfence();  // acquire: subsequent loads see other blocks' h writes
}

// ---------------------------------------------------------------------------
// A-fragment (16x32 bf16, M x K) loaders.
// ISA layout: lanes 0-15 row M=lane&15, elems 0..7 = K(klo..klo+7),
//             elems 8..15 = K(klo+16..klo+23); klo = 0 (lanes<16) else 8.
// p points at &X[row*ld + kb + klo] -> two contiguous 8-elem runs.
// ---------------------------------------------------------------------------
__device__ __forceinline__ v16bf load_a_frag(const bf16* __restrict__ p) {
  v8bf lo = *(const v8bf*)p;          // 16B aligned
  v8bf hi = *(const v8bf*)(p + 16);
  v16bf a;
#pragma unroll
  for (int e = 0; e < 8; ++e) { a[e] = lo[e]; a[8 + e] = hi[e]; }
  return a;
}
__device__ __forceinline__ v16bf load_a_frag(const float* __restrict__ p) {
  v16bf a;
#pragma unroll
  for (int h = 0; h < 2; ++h) {
    const float4* q = (const float4*)(p + h * 16);
    float4 u0 = q[0];
    float4 u1 = q[1];
    a[h * 8 + 0] = (bf16)u0.x; a[h * 8 + 1] = (bf16)u0.y;
    a[h * 8 + 2] = (bf16)u0.z; a[h * 8 + 3] = (bf16)u0.w;
    a[h * 8 + 4] = (bf16)u1.x; a[h * 8 + 5] = (bf16)u1.y;
    a[h * 8 + 6] = (bf16)u1.z; a[h * 8 + 7] = (bf16)u1.w;
  }
  return a;
}

// B-fragment (32x16 bf16, K x N) where B[k][n] = W[n][k], W row-major [3H][K].
// Lane holds column n = row index into W; two 8-elem runs K(klo..+7), K(klo+16..+23), klo = 0/8 (16B aligned).
__device__ __forceinline__ v16bf load_b_w(const bf16* __restrict__ w,
                                          int wrow, int kb, int lane) {
  int klo = (lane < 16) ? 0 : 8;
  const bf16* p = w + (size_t)wrow * HID + kb + klo;
  v8bf lo = *(const v8bf*)p;
  v8bf hi = *(const v8bf*)(p + 16);
  v16bf b;
#pragma unroll
  for (int e = 0; e < 8; ++e) { b[e] = lo[e]; b[8 + e] = hi[e]; }
  return b;
}

// ---------------------------------------------------------------------------
// One GRU layer, one 16x16 h-tile per wave. XT is bf16 or float.
// ---------------------------------------------------------------------------
template <typename XT>
__device__ __forceinline__ void gru_layer_tile(
    const XT* __restrict__ X, int xld,
    const float* __restrict__ Hprev_f, const bf16* __restrict__ Hprev_b,
    const bf16* __restrict__ Wih, const bf16* __restrict__ Whh,
    const float* __restrict__ bih, const float* __restrict__ bhh,
    float* __restrict__ Hnew_f, bf16* __restrict__ Hnew_b,
    float* __restrict__ Out, float* __restrict__ FinalH, int t,
    int m0, int c0, int lane) {
  v8f acc_i[3] = {{0,0,0,0,0,0,0,0},{0,0,0,0,0,0,0,0},{0,0,0,0,0,0,0,0}};
  v8f acc_h[3] = {{0,0,0,0,0,0,0,0},{0,0,0,0,0,0,0,0},{0,0,0,0,0,0,0,0}};

  const int arow = m0 + (lane & 15);
  const int aklo = (lane < 16) ? 0 : 8;
  const int wrow = c0 + (lane & 15);     // weight row = output column
  const XT*   xa = X       + (size_t)arow * xld + aklo;
  const bf16* ha = Hprev_b + (size_t)arow * HID + aklo;

#pragma unroll 2
  for (int kb = 0; kb < HID; kb += 32) {
    v16bf ax = load_a_frag(xa + kb);
    v16bf ah = load_a_frag(ha + kb);
#pragma unroll
    for (int g = 0; g < 3; ++g) {
      v16bf bi = load_b_w(Wih, g * HID + wrow, kb, lane);
      acc_i[g] = __builtin_amdgcn_wmma_f32_16x16x32_bf16(
          false, ax, false, bi, (short)0, acc_i[g], false, false);
      v16bf bh = load_b_w(Whh, g * HID + wrow, kb, lane);
      acc_h[g] = __builtin_amdgcn_wmma_f32_16x16x32_bf16(
          false, ah, false, bh, (short)0, acc_h[g], false, false);
    }
  }

  const int c = c0 + (lane & 15);
  const float bir = bih[c], biz = bih[HID + c], bin = bih[2 * HID + c];
  const float bhr = bhh[c], bhz = bhh[HID + c], bhn = bhh[2 * HID + c];

#pragma unroll
  for (int r = 0; r < 8; ++r) {
    int mr = m0 + ((lane < 16) ? r : 8 + r);   // C/D VGPR layout
    float ir = acc_i[0][r] + bir, iz = acc_i[1][r] + biz, in_ = acc_i[2][r] + bin;
    float hr = acc_h[0][r] + bhr, hz = acc_h[1][r] + bhz, hn  = acc_h[2][r] + bhn;
    float rg = fast_sigmoid(ir + hr);
    float zg = fast_sigmoid(iz + hz);
    float ng = fast_tanh(in_ + rg * hn);
    float hp = Hprev_f[(size_t)mr * HID + c];
    float hv = (1.0f - zg) * ng + zg * hp;
    Hnew_f[(size_t)mr * HID + c] = hv;
    Hnew_b[(size_t)mr * HID + c] = (bf16)hv;   // bf16 shadow: next A-matrix
    if (Out)
      Out[(size_t)mr * SEQ * HID + (size_t)t * HID + c] = hv;
    if (FinalH && t == SEQ - 1)
      FinalH[(size_t)mr * HID + c] = hv;
  }
}

// ---------------------------------------------------------------------------
// Persistent kernel: all 512 steps, grid barriers between layer0/layer1.
// Ping-pong state lives at base + {0, HELEM}: single-base offset arithmetic
// keeps every access a GLOBAL (not FLAT) load/store.
// ---------------------------------------------------------------------------
__global__ void __launch_bounds__(TPB)
gru_persistent(const float* __restrict__ input, const bf16* __restrict__ xbf,
               const float* __restrict__ bih0, const float* __restrict__ bhh0,
               const float* __restrict__ bih1, const float* __restrict__ bhh1,
               const bf16* __restrict__ Wih0, const bf16* __restrict__ Whh0,
               const bf16* __restrict__ Wih1, const bf16* __restrict__ Whh1,
               float* h0f, float* h1f, bf16* h0b, bf16* h1b,
               float* __restrict__ out, float* __restrict__ finalh,
               unsigned* bar) {
  const int lane = threadIdx.x & 31;
  const int wave = threadIdx.x >> 5;
  const int tile = blockIdx.x * (TPB / 32) + wave;  // 0..511
  const int m0 = (tile >> 6) << 4;                  // 8 row-tiles
  const int c0 = (tile & 63) << 4;                  // 64 col-tiles

  for (int t = 0; t < SEQ; ++t) {
    const size_t po = (size_t)(t & 1) * HELEM;       // prev offset
    const size_t qo = (size_t)(1 - (t & 1)) * HELEM; // new offset
    // layer 0: h0_new = GRUCell(x_t, h0_prev)
    if (xbf) {
      gru_layer_tile<bf16>(xbf + (size_t)t * HID, SEQ * HID,
                           h0f + po, h0b + po, Wih0, Whh0, bih0, bhh0,
                           h0f + qo, h0b + qo, nullptr, nullptr, t,
                           m0, c0, lane);
    } else {
      gru_layer_tile<float>(input + (size_t)t * HID, SEQ * HID,
                            h0f + po, h0b + po, Wih0, Whh0, bih0, bhh0,
                            h0f + qo, h0b + qo, nullptr, nullptr, t,
                            m0, c0, lane);
    }
    grid_barrier(bar);
    // layer 1: h1_new = GRUCell(h0_new, h1_prev); emit sequence output
    gru_layer_tile<bf16>(h0b + qo, HID,
                         h1f + po, h1b + po, Wih1, Whh1, bih1, bhh1,
                         h1f + qo, h1b + qo, out, finalh, t,
                         m0, c0, lane);
    grid_barrier(bar);
  }
}

// ---------------------------------------------------------------------------
extern "C" void kernel_launch(void* const* d_in, const int* in_sizes, int n_in,
                              void* d_out, int out_size, void* d_ws, size_t ws_size,
                              hipStream_t stream) {
  const float* input = (const float*)d_in[0];
  // d_in[1] = encoder_out : unused (attention branch skipped in reference)
  const float* enc_h = (const float*)d_in[2];
  const float* W_ih0 = (const float*)d_in[3];
  const float* W_hh0 = (const float*)d_in[4];
  const float* b_ih0 = (const float*)d_in[5];
  const float* b_hh0 = (const float*)d_in[6];
  const float* W_ih1 = (const float*)d_in[7];
  const float* W_hh1 = (const float*)d_in[8];
  const float* b_ih1 = (const float*)d_in[9];
  const float* b_hh1 = (const float*)d_in[10];

  const size_t WELEM = (size_t)3 * HID * HID;   // 3072*1024
  const size_t XELEM = (size_t)BATCH * SEQ * HID;

  char* ws = (char*)d_ws;
  bf16* wb_ih0 = (bf16*)ws;                 ws += WELEM * sizeof(bf16);
  bf16* wb_hh0 = (bf16*)ws;                 ws += WELEM * sizeof(bf16);
  bf16* wb_ih1 = (bf16*)ws;                 ws += WELEM * sizeof(bf16);
  bf16* wb_hh1 = (bf16*)ws;                 ws += WELEM * sizeof(bf16);
  float* h0f = (float*)ws;                  ws += 2 * HELEM * sizeof(float);
  float* h1f = (float*)ws;                  ws += 2 * HELEM * sizeof(float);
  bf16*  h0b = (bf16*)ws;                   ws += 2 * HELEM * sizeof(bf16);
  bf16*  h1b = (bf16*)ws;                   ws += 2 * HELEM * sizeof(bf16);
  unsigned* bar = (unsigned*)ws;            ws += 256;
  bf16* xbf = (bf16*)ws;                    // optional big buffer
  const size_t used = (size_t)(ws - (char*)d_ws);
  const bool have_xbf = ws_size >= used + XELEM * sizeof(bf16);

  float* out    = (float*)d_out;
  float* finalh = out + XELEM;

  // one-time conversions (all sizes are multiples of 8)
  const int wn8 = (int)(WELEM / 8), hn8 = (int)(HELEM / 8);
  cvt_f32_to_bf16_v8<<<(wn8 + 255) / 256, 256, 0, stream>>>(W_ih0, wb_ih0, wn8);
  cvt_f32_to_bf16_v8<<<(wn8 + 255) / 256, 256, 0, stream>>>(W_hh0, wb_hh0, wn8);
  cvt_f32_to_bf16_v8<<<(wn8 + 255) / 256, 256, 0, stream>>>(W_ih1, wb_ih1, wn8);
  cvt_f32_to_bf16_v8<<<(wn8 + 255) / 256, 256, 0, stream>>>(W_hh1, wb_hh1, wn8);
  cvt_f32_to_bf16_v8<<<(hn8 + 255) / 256, 256, 0, stream>>>(enc_h, h0b, hn8);
  cvt_f32_to_bf16_v8<<<(hn8 + 255) / 256, 256, 0, stream>>>(enc_h, h1b, hn8);
  if (have_xbf) {
    const int xn8 = (int)(XELEM / 8);
    cvt_f32_to_bf16_v8<<<(xn8 + 255) / 256, 256, 0, stream>>>(input, xbf, xn8);
  }

  hipMemcpyAsync(h0f, enc_h, HELEM * sizeof(float),
                 hipMemcpyDeviceToDevice, stream);
  hipMemcpyAsync(h1f, enc_h, HELEM * sizeof(float),
                 hipMemcpyDeviceToDevice, stream);
  hipMemsetAsync(bar, 0, 2 * sizeof(unsigned), stream);

  gru_persistent<<<NBLK, TPB, 0, stream>>>(
      input, have_xbf ? xbf : nullptr,
      b_ih0, b_hh0, b_ih1, b_hh1,
      wb_ih0, wb_hh0, wb_ih1, wb_hh1,
      h0f, h1f, h0b, h1b,
      out, finalh, bar);
}
